// GraphConvolution_5179730559509
// MI455X (gfx1250) — matrix operand order repaired and checked by reading, not compiled, not run on hardware
//
#include <hip/hip_runtime.h>
#include <math.h>

// GCNII layer: out = theta*(S@W) + (1-theta)*S + x,  S = (1-a)*spmm(G,x) + a*h0
// MI455X: memory/L2-bound problem; GEMM term done with v_wmma_f32_16x16x32_f16.

typedef __attribute__((ext_vector_type(16))) _Float16 v16h;
typedef __attribute__((ext_vector_type(8)))  _Float16 v8h;
typedef __attribute__((ext_vector_type(8)))  float    v8f;

#define N_NODES 50000
#define FDIM    128
#define KS      (FDIM + 8)   // padded LDS row stride (halves) to break bank alignment

__global__ void zero_ws_kernel(float4* __restrict__ hi4, int n4) {
  int i = blockIdx.x * blockDim.x + threadIdx.x;
  if (i < n4) hi4[i] = make_float4(0.f, 0.f, 0.f, 0.f);
}

// One wave per edge: lane L holds float4 (4 features) of the source row.
__global__ void spmm_scatter_kernel(const float* __restrict__ x,
                                    const int*   __restrict__ erow,
                                    const int*   __restrict__ ecol,
                                    const float* __restrict__ ev,
                                    float*       __restrict__ hi, int E) {
  int e    = blockIdx.x * (blockDim.x >> 5) + (threadIdx.x >> 5);
  int lane = threadIdx.x & 31;
  if (e >= E) return;
  int   r = erow[e];
  int   c = ecol[e];
  float w = ev[e];
  float4 v = ((const float4*)(x + (size_t)c * FDIM))[lane];   // coalesced 512B/row
  float* dst = hi + (size_t)r * FDIM + lane * 4;
  atomicAdd(dst + 0, w * v.x);
  atomicAdd(dst + 1, w * v.y);
  atomicAdd(dst + 2, w * v.z);
  atomicAdd(dst + 3, w * v.w);
}

// Block = 16 output rows x 128 cols; 8 waves, one 16x16 WMMA tile per wave.
__global__ void __launch_bounds__(256)
gcnii_gemm_kernel(const float* __restrict__ x,
                  const float* __restrict__ h0,
                  const float* __restrict__ hi,
                  const float* __restrict__ W,
                  const float* __restrict__ lamda_p,
                  const float* __restrict__ alpha_p,
                  const int*   __restrict__ l_p,
                  float*       __restrict__ out) {
  // Wt: weight transposed to [col][k] halves so B-fragments are contiguous.
  __shared__ __align__(16) _Float16 Wt[FDIM * KS];  // ~34.8 KB
  __shared__ __align__(16) _Float16 As[16 * KS];    // ~4.3 KB  (support, f16)

  const float lamda = lamda_p[0];
  const float alpha = alpha_p[0];
  const float theta = __logf(lamda / (float)l_p[0] + 1.0f);

  const int tid     = threadIdx.x;
  const int rowBase = blockIdx.x * 16;

  // Stage W (f32 global, coalesced) -> f16 LDS transposed.
  for (int i = tid; i < FDIM * FDIM; i += 256) {
    int k = i >> 7, n = i & (FDIM - 1);
    Wt[n * KS + k] = (_Float16)W[i];
  }
  // Stage support tile (f16) for the matrix op.
  for (int i = tid; i < 16 * FDIM; i += 256) {
    int r = i >> 7, k = i & (FDIM - 1);
    size_t g = (size_t)(rowBase + r) * FDIM + k;
    As[r * KS + k] = (_Float16)((1.0f - alpha) * hi[g] + alpha * h0[g]);
  }
  __syncthreads();

  const int lane    = tid & 31;
  const int wave    = tid >> 5;
  const int colBase = wave * 16;
  const int m       = lane & 15;   // A row / B col for this lane
  const int hlf     = lane >> 4;   // half-wave selects K sub-range

  v8f acc = {};
#pragma unroll
  for (int kk = 0; kk < FDIM; kk += 32) {
    // 16-bit A 16x32 layout: lanes 0-15 K={0..7,16..23}, lanes 16-31 K={8..15,24..31}
    v8h alo = *(const v8h*)&As[m * KS + kk + hlf * 8];
    v8h ahi = *(const v8h*)&As[m * KS + kk + 16 + hlf * 8];
    // B 32x16 layout: lanes 0-15 K={0..7,16..23} of col m, lanes 16-31 K={8..15,24..31}
    v8h blo = *(const v8h*)&Wt[(colBase + m) * KS + kk + hlf * 8];
    v8h bhi = *(const v8h*)&Wt[(colBase + m) * KS + kk + 16 + hlf * 8];
    v16h a = __builtin_shufflevector(alo, ahi, 0,1,2,3,4,5,6,7,8,9,10,11,12,13,14,15);
    v16h b = __builtin_shufflevector(blo, bhi, 0,1,2,3,4,5,6,7,8,9,10,11,12,13,14,15);
    // (neg_a, A, neg_b, B, c_mod, C, reuse_a, reuse_b)
    acc = __builtin_amdgcn_wmma_f32_16x16x32_f16(false, a, false, b,
                                                 (short)0, acc, false, false);
  }

  // Epilogue in full f32: recompute support from global (theta is small, so the
  // f16 rounding only touches the theta-scaled GEMM term).
  // C/D layout: VGPR i -> row (i + 8*hlf), col = lane&15.
#pragma unroll
  for (int i = 0; i < 8; ++i) {
    int    r = rowBase + i + 8 * hlf;
    size_t g = (size_t)r * FDIM + colBase + m;
    float sup = (1.0f - alpha) * hi[g] + alpha * h0[g];
    out[g] = theta * acc[i] + (1.0f - theta) * sup + x[g];
  }
}

extern "C" void kernel_launch(void* const* d_in, const int* in_sizes, int n_in,
                              void* d_out, int out_size, void* d_ws, size_t ws_size,
                              hipStream_t stream) {
  (void)n_in; (void)out_size; (void)ws_size;
  const float* x    = (const float*)d_in[0];
  const float* h0   = (const float*)d_in[1];
  const int*   erow = (const int*)d_in[2];
  const int*   ecol = (const int*)d_in[3];
  const float* ev   = (const float*)d_in[4];
  const float* W    = (const float*)d_in[5];
  const float* lam  = (const float*)d_in[6];
  const float* alp  = (const float*)d_in[7];
  const int*   lidx = (const int*)d_in[8];
  float* out = (float*)d_out;
  float* hi  = (float*)d_ws;                 // N*F floats of scratch
  const int E  = in_sizes[2];
  const int NF = N_NODES * FDIM;

  zero_ws_kernel<<<(NF / 4 + 255) / 256, 256, 0, stream>>>((float4*)hi, NF / 4);
  spmm_scatter_kernel<<<(E + 7) / 8, 256, 0, stream>>>(x, erow, ecol, ev, hi, E);
  gcnii_gemm_kernel<<<N_NODES / 16, 256, 0, stream>>>(x, h0, hi, W, lam, alp, lidx, out);
}
